// MoEsparseRouting_80358838108605
// MI455X (gfx1250) — matrix operand order repaired and checked by reading, not compiled, not run on hardware
//
#include <hip/hip_runtime.h>

// ---------------------------------------------------------------------------
// Types for CDNA5 WMMA (wave32): v_wmma_f32_16x16x32_bf16
// ---------------------------------------------------------------------------
typedef __attribute__((ext_vector_type(16))) __bf16          v16bf;
typedef __attribute__((ext_vector_type(8)))  float           v8f;
typedef __attribute__((ext_vector_type(8)))  unsigned short  u16x8;

static constexpr int BATCH = 8;
static constexpr int SEQ   = 2048;
static constexpr int MDIM  = 768;
static constexpr int NDIM  = 768;
static constexpr int RANK  = 16;

// LDS column stride for the staged B slab: 128 B of data (hi 64B + lo 64B)
// padded to 144 B (36 DWORDs). gcd(36,64)=4 -> the 16 lanes of a fragment
// read start at 16 distinct bank-groups of 4 -> conflict-free ds_load_b128.
static constexpr int COLB = 144;

// ---------------------------------------------------------------------------
// bf16 helpers (round-to-nearest-even), no __bf16 arithmetic needed
// ---------------------------------------------------------------------------
__device__ __forceinline__ __bf16 f2bf(float x) {
  unsigned u = __builtin_bit_cast(unsigned, x);
  unsigned r = u + 0x7FFFu + ((u >> 16) & 1u);
  unsigned short h = (unsigned short)(r >> 16);
  return __builtin_bit_cast(__bf16, h);
}
__device__ __forceinline__ float bf2f(__bf16 h) {
  unsigned short s = __builtin_bit_cast(unsigned short, h);
  unsigned u = ((unsigned)s) << 16;
  return __builtin_bit_cast(float, u);
}

// Two 16-byte loads assembled into a 16-element bf16 fragment (global or LDS)
__device__ __forceinline__ v16bf load_frag(const __bf16* p0, const __bf16* p1) {
  union { u16x8 h[2]; v16bf v; } t;
  t.h[0] = *reinterpret_cast<const u16x8*>(p0);
  t.h[1] = *reinterpret_cast<const u16x8*>(p1);
  return t.v;
}

__device__ __forceinline__ v8f wmma_bf16(v16bf a, v16bf b, v8f c) {
  return __builtin_amdgcn_wmma_f32_16x16x32_bf16(
      false, a, false, b, (short)0, c, false, false);
}

// ---------------------------------------------------------------------------
// Kernel 1: per-batch TT factor collapse.
// gates are one-hot -> pick expert e; compute
//   L[m=(ma*96+mb*12+mc)][r3] = sum_{r1,r2} c0[mc,r1] c1[r1,mb,r2] c2[r2,ma,r3]
//   R[r3][n=(na*96+nb*12+nc)] = sum_{r4,r5} c3[r3,na,r4] c4[r4,nb,r5] c5[r5,nc]
// Staged through LDS (A1: 96x16, B1: 16x96).
// ---------------------------------------------------------------------------
__global__ __launch_bounds__(256) void tt_expand_lr(
    const float* __restrict__ gates,
    const float* __restrict__ c0, const float* __restrict__ c1,
    const float* __restrict__ c2, const float* __restrict__ c3,
    const float* __restrict__ c4, const float* __restrict__ c5,
    float* __restrict__ Lw, float* __restrict__ Rw)
{
  const int b = blockIdx.x;
  __shared__ int   s_e;
  __shared__ float A1[96 * 16];   // [(mb*12+mc)][r2]
  __shared__ float B1[16 * 96];   // [r4][(nb*12+nc)]

  if (threadIdx.x == 0) {
    int e = 0; float best = -1.0f;
    for (int i = 0; i < 16; ++i) {
      float g = gates[b * 16 + i];
      if (g > best) { best = g; e = i; }
    }
    s_e = e;
  }
  __syncthreads();
  const int e = s_e;

  // A1[(mb,mc)][r2] = sum_r1 c0[e,0,mc,r1] * c1[e,r1,mb,r2]
  for (int idx = threadIdx.x; idx < 96 * 16; idx += 256) {
    const int r2 = idx & 15, mbc = idx >> 4;
    const int mb = mbc / 12, mc = mbc % 12;
    float s = 0.f;
#pragma unroll
    for (int r1 = 0; r1 < RANK; ++r1)
      s += c0[e * 192 + mc * 16 + r1] * c1[e * 2048 + r1 * 128 + mb * 16 + r2];
    A1[mbc * 16 + r2] = s;
  }
  // B1[r4][(nb,nc)] = sum_r5 c4[e,r4,nb,r5] * c5[e,r5,nc,0]
  for (int idx = threadIdx.x; idx < 16 * 96; idx += 256) {
    const int nbc = idx % 96, r4 = idx / 96;
    const int nb = nbc / 12, nc = nbc % 12;
    float s = 0.f;
#pragma unroll
    for (int r5 = 0; r5 < RANK; ++r5)
      s += c4[e * 2048 + r4 * 128 + nb * 16 + r5] * c5[e * 192 + r5 * 12 + nc];
    B1[r4 * 96 + nbc] = s;
  }
  __syncthreads();

  // L[m][r3] = sum_r2 A1[(mb,mc)][r2] * c2[e,r2,ma,r3]
  float* L = Lw + (size_t)b * MDIM * RANK;
  for (int idx = threadIdx.x; idx < MDIM * RANK; idx += 256) {
    const int r3 = idx & 15, m = idx >> 4;
    const int ma = m / 96, rem = m % 96;
    float s = 0.f;
#pragma unroll
    for (int r2 = 0; r2 < RANK; ++r2)
      s += A1[rem * 16 + r2] * c2[e * 2048 + r2 * 128 + ma * 16 + r3];
    L[m * RANK + r3] = s;
  }
  // R[r3][n] = sum_r4 c3[e,r3,na,r4] * B1[r4][(nb,nc)]
  float* Rr = Rw + (size_t)b * RANK * NDIM;
  for (int idx = threadIdx.x; idx < RANK * NDIM; idx += 256) {
    const int n = idx % NDIM, r3 = idx / NDIM;
    const int na = n / 96, nbc = n % 96;
    float s = 0.f;
#pragma unroll
    for (int r4 = 0; r4 < RANK; ++r4)
      s += c3[e * 2048 + r3 * 128 + na * 16 + r4] * B1[r4 * 96 + nbc];
    Rr[r3 * NDIM + n] = s;
  }
}

// ---------------------------------------------------------------------------
// Kernel 2: combined weight W_b[n][m] = base_w[n][m] + 8 * (L_b R_b)[m][n],
// split into bf16 hi/lo, stored n-major (so B fragments have contiguous K).
// ---------------------------------------------------------------------------
__global__ __launch_bounds__(256) void build_weights(
    const float* __restrict__ base_w,
    const float* __restrict__ Lw, const float* __restrict__ Rw,
    __bf16* __restrict__ Whi, __bf16* __restrict__ Wlo)
{
  const int i = blockIdx.x * 256 + threadIdx.x;   // over BATCH*N*M, n-major
  const int b = i / (NDIM * MDIM);
  const int rem = i % (NDIM * MDIM);
  const int n = rem / MDIM;
  const int m = rem % MDIM;

  const float* L  = Lw + (size_t)b * MDIM * RANK + m * RANK;
  const float* Rr = Rw + (size_t)b * RANK * NDIM;
  float s = 0.f;
#pragma unroll
  for (int r = 0; r < RANK; ++r) s += L[r] * Rr[r * NDIM + n];

  const float w = base_w[n * MDIM + m] + 8.0f * s;   // ALPHA = 8
  const __bf16 hi = f2bf(w);
  const __bf16 lo = f2bf(w - bf2f(hi));
  Whi[i] = hi;
  Wlo[i] = lo;
}

// ---------------------------------------------------------------------------
// Kernel 3: split X (f32) into bf16 hi/lo
// ---------------------------------------------------------------------------
__global__ __launch_bounds__(256) void split_x(
    const float* __restrict__ X, __bf16* __restrict__ Xhi, __bf16* __restrict__ Xlo)
{
  const int i = blockIdx.x * 256 + threadIdx.x;   // over BATCH*SEQ*M
  const float x = X[i];
  const __bf16 hi = f2bf(x);
  Xhi[i] = hi;
  Xlo[i] = f2bf(x - bf2f(hi));
}

// ---------------------------------------------------------------------------
// Kernel 4: batched GEMM, bf16-x3 split with f32 accumulation.
// Block = 256 threads (8 waves) covering a 128(M) x 128(N) macro tile;
// waves arranged 4(M) x 2(N), each wave owns a 32(M) x 64(N) strip.
//
// B slab (128 cols x 32 K, hi+lo) is staged into LDS with the CDNA5 async
// Global->LDS path (GLOBAL_LOAD_ASYNC_TO_LDS_B128, ASYNCcnt), double
// buffered: chunk i+1 streams into LDS while chunk i is consumed by WMMAs.
// Each wave: s_wait_asynccnt 0 + workgroup barrier before switching buffers;
// the next-buffer issue happens after the barrier, so no buffer is written
// while any wave still reads it.
//
// LDS layout per buffer: [col 0..127] x {hi 64B, lo 64B, pad 16B} (COLB=144 B
// per column, bank-conflict-free fragment reads).
// Fragment layouts per CDNA5 ISA 7.12.2:
//   A (16x32 bf16): lane<16 row M=lane, K {k..k+7, k+16..k+23};
//                   lane>=16 row M=lane-16, K {k+8..k+15, k+24..k+31}
//   B (32x16 bf16): lane = column N; lanes<16 K {k..k+7, k+16..k+23}, lanes>=16 K {k+8..k+15, k+24..k+31}
//   C/D f32: element (M = v + 8*(lane>=16), N = lane&15) in VGPR v
// ---------------------------------------------------------------------------
__device__ __forceinline__ void issue_async_b(
    const __bf16* __restrict__ wh, const __bf16* __restrict__ wl,
    int ncolBase, int k, char* bufBase, int tid)
{
  // 1024 x 16B transfers per chunk, 4 per thread (256 threads):
  // idx = [mat:1][col:7][seg:2] ; seg = 16B segment within a 64B column row
#pragma unroll
  for (int j = 0; j < 4; ++j) {
    const int idx = j * 256 + tid;
    const int mat = idx >> 9;          // 0 = hi, 1 = lo
    const int r   = idx & 511;
    const int col = r >> 2;
    const int seg = r & 3;
    const __bf16* src =
        (mat ? wl : wh) + (size_t)(ncolBase + col) * MDIM + k + seg * 8;
    const unsigned dst =
        (unsigned)(uintptr_t)(bufBase + col * COLB + mat * 64 + seg * 16);
    asm volatile("global_load_async_to_lds_b128 %0, %1, off"
                 :
                 : "v"(dst), "v"((unsigned long long)(uintptr_t)src)
                 : "memory");
  }
}

__global__ __launch_bounds__(256) void tt_gemm(
    const __bf16* __restrict__ Xhi, const __bf16* __restrict__ Xlo,
    const __bf16* __restrict__ Whi, const __bf16* __restrict__ Wlo,
    const float* __restrict__ bias, float* __restrict__ out)
{
  __shared__ char Bsh[2][128 * COLB];   // 2 x 18 KB double buffer

  const int b     = blockIdx.z;
  const int wave  = (int)(threadIdx.x >> 5);
  const int mtile = blockIdx.y * 128 + (wave >> 1) * 32;   // 2 row-tiles/wave
  const int nwave = (wave & 1) * 64;                       // wave's N offset in block
  const int ncolBase = blockIdx.x * 128;                   // block's N base
  const int lane  = threadIdx.x & 31;
  const int lrow  = lane & 15;
  const int half  = lane >> 4;          // 0: lanes 0-15, 1: lanes 16-31
  const int tid   = (int)threadIdx.x;

  const size_t arow = (size_t)(b * SEQ + mtile + lrow) * MDIM;
  const __bf16* xh0 = Xhi + arow;
  const __bf16* xl0 = Xlo + arow;
  const __bf16* xh1 = xh0 + (size_t)16 * MDIM;
  const __bf16* xl1 = xl0 + (size_t)16 * MDIM;
  const __bf16* wh  = Whi + (size_t)b * NDIM * MDIM;
  const __bf16* wl  = Wlo + (size_t)b * NDIM * MDIM;

  const v8f zero = {0.f, 0.f, 0.f, 0.f, 0.f, 0.f, 0.f, 0.f};
  v8f acc0[4] = {zero, zero, zero, zero};
  v8f acc1[4] = {zero, zero, zero, zero};

  constexpr int NC = MDIM / 32;        // 24 K-chunks

  // prologue: stream chunk 0 into buffer 0
  issue_async_b(wh, wl, ncolBase, 0, &Bsh[0][0], tid);

  for (int i = 0; i < NC; ++i) {
    const int k = i * 32;

    // my async loads for the current buffer are done...
    asm volatile("s_wait_asynccnt 0" ::: "memory");
    // ...and so are everyone else's (also fences prior-iteration LDS reads)
    __syncthreads();

    // stream next chunk into the other buffer while we compute
    if (i + 1 < NC)
      issue_async_b(wh, wl, ncolBase, k + 32, &Bsh[(i + 1) & 1][0], tid);

    // WGP-scope prefetch of the next A chunk (global_prefetch_b8)
    __builtin_prefetch(xh0 + k + 32, 0, 3);
    __builtin_prefetch(xh1 + k + 32, 0, 3);

    const int ka = k + half * 8;
    const v16bf ahi0 = load_frag(xh0 + ka, xh0 + ka + 16);
    const v16bf alo0 = load_frag(xl0 + ka, xl0 + ka + 16);
    const v16bf ahi1 = load_frag(xh1 + ka, xh1 + ka + 16);
    const v16bf alo1 = load_frag(xl1 + ka, xl1 + ka + 16);

    const char* buf = &Bsh[i & 1][0];
#pragma unroll
    for (int t = 0; t < 4; ++t) {
      const int colLocal = nwave + t * 16 + lrow;
      const char* bcol = buf + colLocal * COLB + half * 16;
      const v16bf bhi = load_frag((const __bf16*)bcol,
                                  (const __bf16*)bcol + 16);
      const v16bf blo = load_frag((const __bf16*)(bcol + 64),
                                  (const __bf16*)(bcol + 64) + 16);
      // hi*hi + hi*lo + lo*hi for both M tiles; B fragment reused 6x
      acc0[t] = wmma_bf16(ahi0, bhi, acc0[t]);
      acc1[t] = wmma_bf16(ahi1, bhi, acc1[t]);
      acc0[t] = wmma_bf16(ahi0, blo, acc0[t]);
      acc1[t] = wmma_bf16(ahi1, blo, acc1[t]);
      acc0[t] = wmma_bf16(alo0, bhi, acc0[t]);
      acc1[t] = wmma_bf16(alo1, bhi, acc1[t]);
    }
  }

#pragma unroll
  for (int t = 0; t < 4; ++t) {
    const int col = ncolBase + nwave + t * 16 + lrow;
    const float bb = bias[col];
#pragma unroll
    for (int v = 0; v < 8; ++v) {
      const int row0 = mtile + v + half * 8;
      out[(size_t)(b * SEQ + row0) * NDIM + col]      = acc0[t][v] + bb;
      out[(size_t)(b * SEQ + row0 + 16) * NDIM + col] = acc1[t][v] + bb;
    }
  }
}

// ---------------------------------------------------------------------------
// Launcher
// ---------------------------------------------------------------------------
extern "C" void kernel_launch(void* const* d_in, const int* in_sizes, int n_in,
                              void* d_out, int out_size, void* d_ws, size_t ws_size,
                              hipStream_t stream) {
  const float* X      = (const float*)d_in[0];
  const float* base_w = (const float*)d_in[1];
  const float* base_b = (const float*)d_in[2];
  const float* gates  = (const float*)d_in[3];
  const float* c0 = (const float*)d_in[4];
  const float* c1 = (const float*)d_in[5];
  const float* c2 = (const float*)d_in[6];
  const float* c3 = (const float*)d_in[7];
  const float* c4 = (const float*)d_in[8];
  const float* c5 = (const float*)d_in[9];

  char* ws = (char*)d_ws;
  size_t off = 0;
  float* Lw = (float*)(ws + off); off += (size_t)BATCH * MDIM * RANK * sizeof(float);
  float* Rw = (float*)(ws + off); off += (size_t)BATCH * RANK * NDIM * sizeof(float);
  __bf16* Whi = (__bf16*)(ws + off); off += (size_t)BATCH * NDIM * MDIM * sizeof(short);
  __bf16* Wlo = (__bf16*)(ws + off); off += (size_t)BATCH * NDIM * MDIM * sizeof(short);
  __bf16* Xhi = (__bf16*)(ws + off); off += (size_t)BATCH * SEQ * MDIM * sizeof(short);
  __bf16* Xlo = (__bf16*)(ws + off); off += (size_t)BATCH * SEQ * MDIM * sizeof(short);

  // 1) collapse TT cores (one-hot gates -> per-batch L, R factors)
  tt_expand_lr<<<BATCH, 256, 0, stream>>>(gates, c0, c1, c2, c3, c4, c5, Lw, Rw);

  // 2) fused weight  W_b = base_w (n-major) + alpha * (L_b R_b)^T   (bf16 hi/lo)
  const int wElems = BATCH * NDIM * MDIM;          // 4,718,592
  build_weights<<<wElems / 256, 256, 0, stream>>>(base_w, Lw, Rw, Whi, Wlo);

  // 3) split activations into bf16 hi/lo
  const int xElems = BATCH * SEQ * MDIM;           // 12,582,912
  split_x<<<xElems / 256, 256, 0, stream>>>(X, Xhi, Xlo);

  // 4) batched WMMA GEMM + bias (B slab async-staged through LDS)
  dim3 grid(NDIM / 128, SEQ / 128, BATCH);         // (6, 16, 8)
  tt_gemm<<<grid, 256, 0, stream>>>(Xhi, Xlo, Whi, Wlo, base_b, (float*)d_out);
}
